// FLALinearAttention_24129126269509
// MI455X (gfx1250) — matrix operand order repaired and checked by reading, not compiled, not run on hardware
//
#include <hip/hip_runtime.h>
#include <hip/hip_bf16.h>

// ---------------------------------------------------------------------------
// CDNA5 (gfx1250) chunked causal linear attention, bf16 WMMA + async LDS copies.
// ---------------------------------------------------------------------------

typedef __attribute__((ext_vector_type(16))) __bf16 v16bf;
typedef __attribute__((ext_vector_type(4)))  __bf16 v4bf;
typedef __attribute__((ext_vector_type(8)))  float  v8f;
typedef __attribute__((ext_vector_type(4)))  int    v4i;

#define L_SEQ   2048
#define HID     1024
#define NHEAD   16
#define HDIM    64
#define CHUNK   64

// ---------------- async global->LDS copy (gfx1250 GLOBAL_LOAD_ASYNC_TO_LDS) --
#if defined(__has_builtin)
#  if __has_builtin(__builtin_amdgcn_global_load_async_to_lds_b128)
#    define HAVE_ASYNC_LDS 1
#  endif
#endif
#ifndef HAVE_ASYNC_LDS
#  define HAVE_ASYNC_LDS 0
#endif

typedef __attribute__((address_space(1))) v4i g1_v4i;   // global v4i
typedef __attribute__((address_space(3))) v4i l3_v4i;   // LDS v4i

__device__ __forceinline__ void copy16_g2l(void* lds_dst, const void* gsrc) {
#if HAVE_ASYNC_LDS
    __builtin_amdgcn_global_load_async_to_lds_b128(
        (g1_v4i*)gsrc, (l3_v4i*)lds_dst, 0, 0);
#else
    *(uint4*)lds_dst = *(const uint4*)gsrc;
#endif
}

__device__ __forceinline__ void copy_wait() {
#if HAVE_ASYNC_LDS
#  if __has_builtin(__builtin_amdgcn_s_wait_asynccnt)
    __builtin_amdgcn_s_wait_asynccnt(0);
#  else
    asm volatile("s_wait_asynccnt 0" ::: "memory");
#  endif
#endif
}

// ---------------- WMMA helpers ---------------------------------------------
__device__ __forceinline__ v8f wmma_bf16(v16bf a, v16bf b, v8f c) {
    return __builtin_amdgcn_wmma_f32_16x16x32_bf16(
        false, a, false, b, (short)0, c, false, false);
}

// A-fragment 16x32 bf16, source [M][K] row-major (stride ld).
__device__ __forceinline__ v16bf load_frag_a(const __bf16* base, int row0, int ld, int lane) {
    const __bf16* p = base + (size_t)(row0 + (lane & 15)) * ld;
    const int kb = (lane < 16) ? 0 : 16;
    v16bf f;
#pragma unroll
    for (int i = 0; i < 16; ++i) f[i] = p[kb + i];
    return f;
}

// A-fragment from transposed source: logical A[m][k] = base[k][m].
__device__ __forceinline__ v16bf load_frag_at(const __bf16* base, int row0, int ld, int lane) {
    const int m = row0 + (lane & 15);
    const int kb = (lane < 16) ? 0 : 16;
    v16bf f;
#pragma unroll
    for (int i = 0; i < 16; ++i)
        f[i] = base[(size_t)(kb + i) * ld + m];
    return f;
}

// B-fragment 32x16 bf16, source [K][N] row-major.
__device__ __forceinline__ v16bf load_frag_b(const __bf16* base, int col0, int ld, int lane) {
    const int n = col0 + (lane & 15);
    const int kb = (lane < 16) ? 0 : 16;
    v16bf f;
#pragma unroll
    for (int i = 0; i < 16; ++i) f[i] = base[(size_t)(kb + i) * ld + n];
    return f;
}

// B-fragment from transposed source: logical B[k][n] = base[n][k].
__device__ __forceinline__ v16bf load_frag_bt(const __bf16* base, int col0, int ld, int lane) {
    const __bf16* p = base + (size_t)(col0 + (lane & 15)) * ld;
    const int kb = (lane < 16) ? 0 : 16;
    v16bf f;
#pragma unroll
    for (int i = 0; i < 16; ++i) f[i] = p[kb + i];
    return f;
}

// ---------------- f32 -> bf16 convert (4-wide) ------------------------------
__global__ __launch_bounds__(256) void cvt_bf16_kernel(const float* __restrict__ in,
                                                       __bf16* __restrict__ out, int n4) {
    int i = blockIdx.x * 256 + threadIdx.x;
    if (i < n4) {
        float4 f = ((const float4*)in)[i];
        v4bf o;
        o[0] = (__bf16)f.x; o[1] = (__bf16)f.y; o[2] = (__bf16)f.z; o[3] = (__bf16)f.w;
        *(v4bf*)(out + (size_t)i * 4) = o;
    }
}

// ---------------------------------------------------------------------------
// NT GEMM: out[M,N] = A[M,K] * B[N,K]^T  (bf16 in, f32 accumulate)
// Workgroup tile 128x128, 8 waves (4M x 2N), each wave 32x64 = 2x4 tiles.
// Double-buffered LDS, async global->LDS staging.
// ACT: elu(x)+1 ; OUTF: write f32 instead of bf16.
// ---------------------------------------------------------------------------
#define GTM 128
#define GTN 128
#define GTK 32
#define GLD (GTK + 8)

template <int ACT, int OUTF>
__global__ __launch_bounds__(256) void gemm_nt_kernel(
    const __bf16* __restrict__ A, const __bf16* __restrict__ B,
    __bf16* __restrict__ outb, float* __restrict__ outf,
    int M, int N, int K)
{
    __shared__ __bf16 As[2][GTM * GLD];
    __shared__ __bf16 Bs[2][GTN * GLD];

    const int tid  = threadIdx.x;
    const int lane = tid & 31;
    const int wave = tid >> 5;              // wave32
    const int m0 = blockIdx.y * GTM;
    const int n0 = blockIdx.x * GTN;
    const int wm = (wave & 3) * 32;         // 4 waves along M
    const int wn = (wave >> 2) * 64;        // 2 waves along N

    v8f acc[2][4] = {};

    // stage one K-slice (128x32 A + 128x32 B) into LDS buffer `buf`
    auto stage = [&](int buf, int k0) {
#pragma unroll
        for (int u = 0; u < 2; ++u) {       // 512 x 16B chunks for A
            int idx = tid + u * 256;
            int r = idx >> 2, c = (idx & 3) * 8;
            copy16_g2l(&As[buf][r * GLD + c], A + (size_t)(m0 + r) * K + k0 + c);
        }
#pragma unroll
        for (int u = 0; u < 2; ++u) {       // 512 x 16B chunks for B
            int idx = tid + u * 256;
            int r = idx >> 2, c = (idx & 3) * 8;
            copy16_g2l(&Bs[buf][r * GLD + c], B + (size_t)(n0 + r) * K + k0 + c);
        }
    };

    stage(0, 0);
    copy_wait();
    __syncthreads();

    int buf = 0;
    for (int k0 = 0; k0 < K; k0 += GTK, buf ^= 1) {
        if (k0 + GTK < K) stage(buf ^ 1, k0 + GTK);   // overlap next-tile copies

        v16bf fb[4];
#pragma unroll
        for (int j = 0; j < 4; ++j)
            fb[j] = load_frag_bt(Bs[buf], wn + j * 16, GLD, lane);
#pragma unroll
        for (int i = 0; i < 2; ++i) {
            v16bf fa = load_frag_a(As[buf], wm + i * 16, GLD, lane);
#pragma unroll
            for (int j = 0; j < 4; ++j)
                acc[i][j] = wmma_bf16(fa, fb[j], acc[i][j]);
        }

        copy_wait();          // next-buffer copies done (this wave)
        __syncthreads();      // all waves done
    }

    // epilogue (branchless activation, compile-time output type)
#pragma unroll
    for (int i = 0; i < 2; ++i) {
#pragma unroll
        for (int j = 0; j < 4; ++j) {
            const int ng = n0 + wn + j * 16 + (lane & 15);
#pragma unroll
            for (int r = 0; r < 8; ++r) {
                const int mg = m0 + wm + i * 16 + r + ((lane >= 16) ? 8 : 0);
                float v = acc[i][j][r];
                if (ACT) {
                    float e = __expf(fminf(v, 0.f));     // elu(x)+1
                    v = (v > 0.f) ? (v + 1.f) : e;
                }
                if (OUTF) outf[(size_t)mg * N + ng] = v;
                else      outb[(size_t)mg * N + ng] = (__bf16)v;
            }
        }
    }
}

// ---------------------------------------------------------------------------
// Chunked causal linear-attention scan.
// grid = NHEAD*4 (head h, 16-wide e-column block); block = 256 (8 waves).
// Per chunk c (C=64):
//   A    = causal_mask(Q K^T)                    (16 tiles over 8 waves)
//   dnm  = rowsum(A) + Q . z_prev                (64 threads)
//   out  = (A V + Q S_prev) / dnm                (waves 0-3)
//   S   += K^T V ; z += colsum(K)                (waves 4-7 / 64 threads)
// S kept f32 in LDS, bf16 snapshot per chunk for WMMA.
// ---------------------------------------------------------------------------
#define ALD  72   // bf16 row stride, 64-wide tiles (144B, 16B aligned)
#define VLD  24   // bf16 row stride, 16-wide tiles (48B, 16B aligned)
#define SLD  20   // f32 row stride, 16-wide state

__global__ __launch_bounds__(256) void linattn_chunk_kernel(
    const __bf16* __restrict__ qg, const __bf16* __restrict__ kg,
    const __bf16* __restrict__ vg, __bf16* __restrict__ og)
{
    __shared__ __bf16 Qs[CHUNK * ALD];
    __shared__ __bf16 Ks[CHUNK * ALD];
    __shared__ __bf16 As[CHUNK * ALD];
    __shared__ __bf16 Vs[CHUNK * VLD];
    __shared__ __bf16 Sb[HDIM * VLD];
    __shared__ float  Ss[HDIM * SLD];
    __shared__ float  zs[HDIM];
    __shared__ float  dnm[CHUNK];

    const int tid  = threadIdx.x;
    const int lane = tid & 31;
    const int wave = tid >> 5;
    const int h  = blockIdx.x >> 2;
    const int eb = blockIdx.x & 3;

    for (int i = tid; i < HDIM * SLD; i += 256) Ss[i] = 0.f;
    if (tid < HDIM) zs[tid] = 0.f;
    __syncthreads();

    for (int c = 0; c < L_SEQ / CHUNK; ++c) {
        const size_t rowbase = (size_t)c * CHUNK;

        // ---- async-stage Q,K (64x64) and V e-block (64x16); snapshot S->bf16
#pragma unroll
        for (int u = 0; u < 2; ++u) {            // 512 x 16B chunks each
            int idx = tid + u * 256;
            int r = idx >> 3, cc = (idx & 7) * 8;
            size_t gb = (rowbase + r) * HID + (size_t)h * HDIM + cc;
            copy16_g2l(&Qs[r * ALD + cc], qg + gb);
            copy16_g2l(&Ks[r * ALD + cc], kg + gb);
        }
        if (tid < 128) {                          // 128 x 16B chunks for V
            int r = tid >> 1, cc = (tid & 1) * 8;
            copy16_g2l(&Vs[r * VLD + cc],
                       vg + (rowbase + r) * HID + (size_t)h * HDIM + eb * 16 + cc);
        }
        for (int idx = tid; idx < HDIM * 16; idx += 256) {
            int r = idx >> 4, e = idx & 15;
            Sb[r * VLD + e] = (__bf16)Ss[r * SLD + e];
        }
        copy_wait();
        __syncthreads();

        // ---- Stage A: A = causal_mask(Q K^T)
#pragma unroll
        for (int t = 0; t < 2; ++t) {
            const int tile = wave * 2 + t;
            const int tm = (tile >> 2) * 16, tn = (tile & 3) * 16;
            v8f a = {};
#pragma unroll
            for (int ks = 0; ks < HDIM; ks += 32) {
                v16bf fa = load_frag_a(Qs + ks, tm, ALD, lane);
                v16bf fb = load_frag_bt(Ks + ks, tn, ALD, lane);
                a = wmma_bf16(fa, fb, a);
            }
            const int nl = tn + (lane & 15);
            const int mb = tm + ((lane >= 16) ? 8 : 0);
#pragma unroll
            for (int r = 0; r < 8; ++r) {
                const int ml = mb + r;
                As[ml * ALD + nl] = (__bf16)((nl <= ml) ? a[r] : 0.f);
            }
        }
        __syncthreads();

        // ---- Stage B: denominators
        if (tid < CHUNK) {
            float s = 0.f;
            for (int n = 0; n < CHUNK; ++n) s += (float)As[tid * ALD + n];
            for (int d = 0; d < HDIM; ++d)  s += (float)Qs[tid * ALD + d] * zs[d];
            dnm[tid] = fmaxf(s, 1e-6f);
        }
        __syncthreads();

        // ---- Stage C
        if (wave < 4) {
            const int tm = wave * 16;
            v8f o1 = {}, o2 = {};                 // split accumulators
#pragma unroll
            for (int ks = 0; ks < HDIM; ks += 32) {
                v16bf fa = load_frag_a(As + ks, tm, ALD, lane);
                v16bf fv = load_frag_b(Vs + ks * VLD, 0, VLD, lane);
                o1 = wmma_bf16(fa, fv, o1);
                v16bf fq = load_frag_a(Qs + ks, tm, ALD, lane);
                v16bf fs = load_frag_b(Sb + ks * VLD, 0, VLD, lane);
                o2 = wmma_bf16(fq, fs, o2);
            }
            const int nl = lane & 15;
            const int mb = tm + ((lane >= 16) ? 8 : 0);
#pragma unroll
            for (int r = 0; r < 8; ++r) {
                const int ml = mb + r;
                float v = (o1[r] + o2[r]) / dnm[ml];
                og[(rowbase + ml) * HID + (size_t)h * HDIM + eb * 16 + nl] = (__bf16)v;
            }
        } else {
            const int td = (wave - 4) * 16;
            v8f su = {};
#pragma unroll
            for (int ks = 0; ks < HDIM; ks += 32) {
                v16bf fk = load_frag_at(Ks + ks * ALD, td, ALD, lane);
                v16bf fv = load_frag_b(Vs + ks * VLD, 0, VLD, lane);
                su = wmma_bf16(fk, fv, su);
            }
            const int nl = lane & 15;
            const int db = td + ((lane >= 16) ? 8 : 0);
#pragma unroll
            for (int r = 0; r < 8; ++r)
                Ss[(db + r) * SLD + nl] += su[r];
        }
        if (tid < HDIM) {
            float z = zs[tid];
            for (int n = 0; n < CHUNK; ++n) z += (float)Ks[n * ALD + tid];
            zs[tid] = z;
        }
        __syncthreads();
    }
}

// ---------------------------------------------------------------------------
// Launch
// ---------------------------------------------------------------------------
extern "C" void kernel_launch(void* const* d_in, const int* in_sizes, int n_in,
                              void* d_out, int out_size, void* d_ws, size_t ws_size,
                              hipStream_t stream)
{
    (void)in_sizes; (void)n_in; (void)out_size; (void)ws_size;
    const float* x  = (const float*)d_in[0];
    const float* Wq = (const float*)d_in[1];
    const float* Wk = (const float*)d_in[2];
    const float* Wv = (const float*)d_in[3];
    const float* Wo = (const float*)d_in[4];
    float* out = (float*)d_out;

    char* ws = (char*)d_ws;
    const size_t MB = 1024 * 1024;
    __bf16* xb  = (__bf16*)(ws + 0 * MB);
    __bf16* Wqb = (__bf16*)(ws + 4 * MB);
    __bf16* Wkb = (__bf16*)(ws + 6 * MB);
    __bf16* Wvb = (__bf16*)(ws + 8 * MB);
    __bf16* Wob = (__bf16*)(ws + 10 * MB);
    __bf16* qb  = (__bf16*)(ws + 12 * MB);
    __bf16* kb  = (__bf16*)(ws + 16 * MB);
    __bf16* vb  = (__bf16*)(ws + 20 * MB);
    __bf16* ab  = (__bf16*)(ws + 24 * MB);

    const int NX4 = L_SEQ * HID / 4;
    const int NW4 = HID * HID / 4;
    cvt_bf16_kernel<<<(NX4 + 255) / 256, 256, 0, stream>>>(x,  xb,  NX4);
    cvt_bf16_kernel<<<(NW4 + 255) / 256, 256, 0, stream>>>(Wq, Wqb, NW4);
    cvt_bf16_kernel<<<(NW4 + 255) / 256, 256, 0, stream>>>(Wk, Wkb, NW4);
    cvt_bf16_kernel<<<(NW4 + 255) / 256, 256, 0, stream>>>(Wv, Wvb, NW4);
    cvt_bf16_kernel<<<(NW4 + 255) / 256, 256, 0, stream>>>(Wo, Wob, NW4);

    dim3 g(HID / GTN, L_SEQ / GTM);   // (8, 16)
    gemm_nt_kernel<1, 0><<<g, 256, 0, stream>>>(xb, Wqb, qb, nullptr, L_SEQ, HID, HID);
    gemm_nt_kernel<1, 0><<<g, 256, 0, stream>>>(xb, Wkb, kb, nullptr, L_SEQ, HID, HID);
    gemm_nt_kernel<0, 0><<<g, 256, 0, stream>>>(xb, Wvb, vb, nullptr, L_SEQ, HID, HID);

    linattn_chunk_kernel<<<NHEAD * 4, 256, 0, stream>>>(qb, kb, vb, ab);

    gemm_nt_kernel<0, 1><<<g, 256, 0, stream>>>(ab, Wob, nullptr, out, L_SEQ, HID, HID);
}
